// VanillaRNN_32959579030293
// MI455X (gfx1250) — matrix operand order repaired and checked by reading, not compiled, not run on hardware
//
#include <hip/hip_runtime.h>

// ---------------------------------------------------------------------------
// VanillaRNN on MI455X (gfx1250, wave32, WMMA, persistent-LDS)
//   S=512, B=256, E=256, H=1024, C=10
//
//   table[10,H] = emb @ W_hx + b_h        (tiny; replaces 68-GFLOP gather GEMM)
//   state       = tanh(table[x[s]] + state @ W_hh)  x512   (275 GFLOP, WMMA)
//   out[B,C]    = state @ W_ph + b_p      (tiny)
//
// Persistent kernel: 64 blocks x 512 threads, one block per WGP.
// Block (mb,nb) owns a 64x64 output tile; its 128KB W_hh slab lives in LDS
// for all 512 steps; the 64-row state slab is re-staged into padded LDS each
// step; steps are separated by a device-wide atomic barrier.
// ---------------------------------------------------------------------------

typedef __attribute__((ext_vector_type(16))) __bf16 v16bf;
typedef __attribute__((ext_vector_type(8)))  float  v8f;
typedef __attribute__((ext_vector_type(4)))  int    v4i;

#define SDIM 512
#define BDIM 256
#define EDIM 256
#define HDIM 1024
#define CDIM 10

#define NBLK   64          // 4 M-groups x 16 N-groups
#define NTHR   512         // 16 waves
#define AROWP  1032        // padded LDS row stride (bf16): 516 dwords, 516%64==4
#define B_LDS_ELEMS (32 * 2048)               // 65536 bf16 = 128 KB
#define A_LDS_ELEMS (64 * AROWP)              // 66048 bf16 = 129 KB
#define SMEM_BYTES ((B_LDS_ELEMS + A_LDS_ELEMS) * 2)

union BfFrag { v16bf v; v4i q[2]; };

// ---------------------------------------------------------------------------
// Pack W_hh (f32 row-major [K=H][N=H]) into bf16 WMMA-B fragment order:
// idx = ((kb*(H/16)+nb)*32 + lane)*16 + j ; lane<16 -> N=nb*16+lane, K=kb*32+j
// lane>=16 -> N=nb*16+lane-16, K=kb*32+16+j. 16 contiguous bf16 per lane.
// ---------------------------------------------------------------------------
__global__ void pack_whh_kernel(const float* __restrict__ W, __bf16* __restrict__ Wp) {
    int t = blockIdx.x * blockDim.x + threadIdx.x;      // H*H threads
    int j    = t & 15;
    int lane = (t >> 4) & 31;
    int nb   = (t >> 9) & (HDIM / 16 - 1);
    int kb   = t >> 15;
    int K = kb * 32 + ((lane >> 4) << 4) + j;
    int N = nb * 16 + (lane & 15);
    Wp[t] = (__bf16)W[K * HDIM + N];
}

// table[v][h] = sum_e emb[v][e] * W_hx[e][h] + b_h[h]   (10 x 1024)
__global__ void make_table_kernel(const float* __restrict__ emb,
                                  const float* __restrict__ W_hx,
                                  const float* __restrict__ b_h,
                                  float* __restrict__ table) {
    int t = blockIdx.x * blockDim.x + threadIdx.x;
    int v = t / HDIM, h = t % HDIM;
    float acc = b_h[h];
    const float* er = emb + v * EDIM;
    for (int e = 0; e < EDIM; ++e) acc += er[e] * W_hx[e * HDIM + h];
    table[t] = acc;
}

// Reset barrier state (workspace is poisoned 0xAA; must re-init every call).
__global__ void init_barrier_kernel(unsigned* __restrict__ bar) {
    bar[0] = 0u;   // arrival count
    bar[1] = 0u;   // generation
}

// Device-wide barrier: generation counter + agent-scope atomics + s_sleep.
__device__ __forceinline__ void grid_barrier(unsigned* cnt, unsigned* gen) {
    __syncthreads();
    if (threadIdx.x == 0) {
        __threadfence();   // make this block's global stores visible (L2)
        unsigned g = __hip_atomic_load(gen, __ATOMIC_RELAXED,
                                       __HIP_MEMORY_SCOPE_AGENT);
        unsigned arrived = __hip_atomic_fetch_add(cnt, 1u, __ATOMIC_ACQ_REL,
                                                  __HIP_MEMORY_SCOPE_AGENT) + 1u;
        if (arrived == (unsigned)NBLK) {
            __hip_atomic_store(cnt, 0u, __ATOMIC_RELAXED,
                               __HIP_MEMORY_SCOPE_AGENT);
            __hip_atomic_fetch_add(gen, 1u, __ATOMIC_RELEASE,
                                   __HIP_MEMORY_SCOPE_AGENT);
        } else {
            while (__hip_atomic_load(gen, __ATOMIC_ACQUIRE,
                                     __HIP_MEMORY_SCOPE_AGENT) == g)
                __builtin_amdgcn_s_sleep(2);
        }
    }
    __syncthreads();
}

// ---------------------------------------------------------------------------
// Persistent recurrence kernel.
// ---------------------------------------------------------------------------
__global__ __launch_bounds__(NTHR) void rnn_persistent_kernel(
    const __bf16* __restrict__ Wp,     // packed W_hh bf16 (2 MB, L2-resident)
    const float*  __restrict__ table,  // [10][H] f32
    const int*    __restrict__ x,      // [S][B] class ids
    __bf16*       __restrict__ sbuf0,  // [B][H] state ping
    __bf16*       __restrict__ sbuf1,  // [B][H] state pong
    unsigned*     __restrict__ bar)    // barrier {count, gen}
{
    extern __shared__ char smem[];
    __bf16* Bsl = (__bf16*)smem;                              // [32][4*512]
    __bf16* Asl = (__bf16*)(smem + B_LDS_ELEMS * 2);          // [64][AROWP]

    const int tid  = threadIdx.x;
    const int lane = tid & 31;
    const int w    = tid >> 5;          // wave 0..15
    const int mb   = blockIdx.x >> 4;   // 0..3   (64-row group)
    const int nbg  = blockIdx.x & 15;   // 0..15  (64-col group)
    const int nb0  = nbg * 4;
    const int rowbase = mb * 64;

    // ---- Stage this block's W_hh slab into LDS once (128 KB, resident). ----
#pragma unroll
    for (int it = 0; it < 16; ++it) {
        int cc  = tid + it * NTHR;          // 0..8191 chunks of 8 bf16
        int kb  = cc >> 8;                  // 0..31
        int rem = cc & 255;
        v4i v = *(const v4i*)(Wp + (size_t)(kb * 64 + nb0) * 512 + rem * 8);
        *(v4i*)(Bsl + kb * 2048 + rem * 8) = v;
    }

    // ---- Zero initial state (workspace is poisoned): 1 x b128 per thread. --
    {
        v4i z = {};
        ((v4i*)sbuf0)[blockIdx.x * NTHR + tid] = z;
    }
    grid_barrier(bar, bar + 1);

    // Per-wave tile: rows rowbase + tm*16 .. +15, cols nbg*64 + tn*16 .. +15.
    const int tm = w >> 2, tn = w & 3;
    const __bf16* albase = Asl + (size_t)(tm * 16 + (lane & 15)) * AROWP
                               + ((lane >> 4) << 4);
    const __bf16* blbase = Bsl + tn * 512 + lane * 16;
    const int colg  = nbg * 64 + tn * 16 + (lane & 15);
    const int rhalf = (lane >> 4) << 3;

    for (int s = 0; s < SDIM; ++s) {
        const __bf16* sin  = (s & 1) ? sbuf1 : sbuf0;
        __bf16*       sout = (s & 1) ? sbuf0 : sbuf1;

        // ---- Stage 64-row A slab into padded LDS (128 KB). ----
#pragma unroll
        for (int it = 0; it < 16; ++it) {
            int cc  = tid + it * NTHR;      // 0..8191 chunks of 8 bf16
            int row = cc >> 7;              // 0..63
            int c8  = cc & 127;
            v4i v = *(const v4i*)(sin + (size_t)(rowbase + row) * HDIM + c8 * 8);
            *(v4i*)(Asl + row * AROWP + c8 * 8) = v;
        }
        __syncthreads();

        // ---- 16x16x1024 from LDS: 32 x v_wmma_f32_16x16x32_bf16. ----
        v8f c = {};
#pragma unroll
        for (int kb = 0; kb < HDIM / 32; ++kb) {
            BfFrag a, b;
            a.q[0] = *(const v4i*)(albase + kb * 32);
            a.q[1] = *(const v4i*)(albase + kb * 32 + 8);
            b.q[0] = *(const v4i*)(blbase + kb * 2048);
            b.q[1] = *(const v4i*)(blbase + kb * 2048 + 8);
            c = __builtin_amdgcn_wmma_f32_16x16x32_bf16(false, a.v, false, b.v,
                                                        (short)0, c, false, false);
        }

        // ---- Fused epilogue: x_proj gather + tanh + bf16 store. ----
        const int* xrow = x + (size_t)s * BDIM;
#pragma unroll
        for (int r = 0; r < 8; ++r) {
            int rowg = rowbase + tm * 16 + rhalf + r;
            const float* trow = table + (size_t)xrow[rowg] * HDIM;
            sout[(size_t)rowg * HDIM + colg] = (__bf16)tanhf(c[r] + trow[colg]);
        }

        grid_barrier(bar, bar + 1);   // all writes of step s visible
    }
}

// out[b][c] = sum_h state[b][h] * W_ph[h][c] + b_p[c]   (256 x 10, tiny)
__global__ void proj_kernel(const __bf16* __restrict__ s,
                            const float* __restrict__ W_ph,
                            const float* __restrict__ b_p,
                            float* __restrict__ out) {
    int t = blockIdx.x * blockDim.x + threadIdx.x;
    if (t >= BDIM * CDIM) return;
    int b = t / CDIM, c = t % CDIM;
    float acc = b_p[c];
    const __bf16* sr = s + (size_t)b * HDIM;
    for (int h = 0; h < HDIM; ++h)
        acc += (float)sr[h] * W_ph[h * CDIM + c];
    out[t] = acc;
}

// ---------------------------------------------------------------------------
extern "C" void kernel_launch(void* const* d_in, const int* in_sizes, int n_in,
                              void* d_out, int out_size, void* d_ws, size_t ws_size,
                              hipStream_t stream) {
    (void)in_sizes; (void)n_in; (void)out_size; (void)ws_size;
    const int*   x    = (const int*)  d_in[0];   // [S,B]
    const float* emb  = (const float*)d_in[1];   // [10,E]
    const float* W_hx = (const float*)d_in[2];   // [E,H]
    const float* W_hh = (const float*)d_in[3];   // [H,H]
    const float* W_ph = (const float*)d_in[4];   // [H,C]
    const float* b_h  = (const float*)d_in[5];   // [H]
    const float* b_p  = (const float*)d_in[6];   // [C]

    char* ws = (char*)d_ws;
    __bf16*   Wp    = (__bf16*)ws;                                        // 2 MB
    float*    table = (float*)(ws + (size_t)2 * 1024 * 1024);             // 40 KB
    __bf16*   s0    = (__bf16*)(ws + (size_t)2 * 1024 * 1024 + 64 * 1024);
    __bf16*   s1    = s0 + (size_t)BDIM * HDIM;                           // 2x512KB
    unsigned* bar   = (unsigned*)((char*)(s1 + (size_t)BDIM * HDIM) + 256);

    pack_whh_kernel   <<<(HDIM * HDIM) / 256, 256, 0, stream>>>(W_hh, Wp);
    make_table_kernel <<<(10 * HDIM) / 256,   256, 0, stream>>>(emb, W_hx, b_h, table);
    init_barrier_kernel<<<1, 1, 0, stream>>>(bar);

    rnn_persistent_kernel<<<NBLK, NTHR, SMEM_BYTES, stream>>>(Wp, table, x,
                                                              s0, s1, bar);

    // 512 steps: final state lands in s0.
    proj_kernel<<<(BDIM * CDIM + 255) / 256, 256, 0, stream>>>(s0, W_ph, b_p,
                                                               (float*)d_out);
}
